// DecorrelateLossClass_79224966742870
// MI455X (gfx1250) — compile-verified, matrix operand order checked by reading, not run on hardware
//
#include <hip/hip_runtime.h>

// ---------------------------------------------------------------------------
// DecorrelateLoss on MI455X (gfx1250, wave32, WMMA)
//   x: [B=262144, C=128] f32, y: [B] int32 in [0,128)
//   out: scalar f32
// Pipeline (all deterministic, no float atomics):
//   1. k_hist    : per-block class histograms (256 blocks x 1024 rows)
//   2. k_scan    : exclusive scan over blocks per class -> counts + offsets
//   3. k_scatter : counting-sort row indices per class (stable, first-rows)
//   4. k_stats   : per-class mean / rstd (full counts, cap 4096)
//   5. k_gemm    : per-class Gram via v_wmma_f32_16x16x32_f16, off-diag^2 sum
//   6. k_final   : fixed-order reduction -> loss
// ---------------------------------------------------------------------------

#define B_ROWS   262144
#define C_DIM    128
#define KCLS     128
#define NB_HIST  256            // histogram / scatter blocks
#define RPB      1024           // rows per histogram / scatter block
#define PADIDX   4096           // stats row cap (>> max binomial count)
#define PADG     2048           // Gram row cap (reference PAD)
#define LDS_STRIDE 72           // halves per staged column row; 144B, 16B-aligned
#define EPSV     1e-8f

typedef __attribute__((ext_vector_type(16))) _Float16 v16h;
typedef __attribute__((ext_vector_type(8)))  _Float16 v8h;
typedef __attribute__((ext_vector_type(8)))  float    v8f;

// ---------------- 1. per-block histogram ----------------
__global__ __launch_bounds__(256)
void k_hist(const int* __restrict__ y, int* __restrict__ blockHist) {
  __shared__ int h[KCLS];
  const int t = threadIdx.x;
  if (t < KCLS) h[t] = 0;
  __syncthreads();
  const int base = blockIdx.x * RPB;
  #pragma unroll
  for (int i = 0; i < RPB / 256; ++i) {
    atomicAdd(&h[y[base + t + i * 256]], 1);     // integer LDS atomics: deterministic
  }
  __syncthreads();
  if (t < KCLS) blockHist[blockIdx.x * KCLS + t] = h[t];
}

// ---------------- 2. per-class exclusive scan over blocks ----------------
__global__ __launch_bounds__(128)
void k_scan(const int* __restrict__ blockHist, int* __restrict__ blockOffset,
            int* __restrict__ counts) {
  const int k = threadIdx.x;
  if (k >= KCLS) return;
  int run = 0;
  for (int b = 0; b < NB_HIST; ++b) {
    blockOffset[b * KCLS + k] = run;
    run += blockHist[b * KCLS + k];
  }
  counts[k] = run;
}

// ---------------- 3. stable counting-sort scatter of row indices ----------------
__global__ __launch_bounds__(256)
void k_scatter(const int* __restrict__ y, const int* __restrict__ blockOffset,
               int* __restrict__ idxbuf) {
  __shared__ int ly[RPB];
  const int t = threadIdx.x;
  const int base = blockIdx.x * RPB;
  #pragma unroll
  for (int i = 0; i < RPB / 256; ++i) ly[t + i * 256] = y[base + t + i * 256];
  __syncthreads();
  #pragma unroll
  for (int i = 0; i < RPB / 256; ++i) {
    const int j   = t + i * 256;
    const int cls = ly[j];
    int rank = 0;
    for (int q = 0; q < j; ++q) rank += (ly[q] == cls);   // deterministic in-block rank
    const int pos = blockOffset[blockIdx.x * KCLS + cls] + rank;
    if (pos < PADIDX) idxbuf[cls * PADIDX + pos] = base + j;
  }
}

// ---------------- 4. per-class mean / rstd ----------------
__global__ __launch_bounds__(256)
void k_stats(const float* __restrict__ x, const int* __restrict__ idxbuf,
             const int* __restrict__ counts,
             float* __restrict__ meanw, float* __restrict__ rstdw) {
  const int k  = blockIdx.x;
  const int t  = threadIdx.x;
  const int c  = t & (C_DIM - 1);
  const int rh = t >> 7;                 // 0/1: interleave rows between thread halves
  const int cnt = counts[k];
  const int P   = (cnt < PADIDX) ? cnt : PADIDX;
  float s = 0.f, ss = 0.f;
  for (int p = rh; p < P; p += 2) {
    if (p + 8 < P) __builtin_prefetch(&idxbuf[k * PADIDX + p + 8], 0, 1);
    const int row = idxbuf[k * PADIDX + p];
    const float v = x[row * C_DIM + c];
    s += v; ss += v * v;
  }
  __shared__ float sh[256], sh2[256];
  sh[t] = s; sh2[t] = ss;
  __syncthreads();
  if (rh == 0) {
    const float st  = sh[t]  + sh[t + 128];
    const float sst = sh2[t] + sh2[t + 128];
    const float fc  = (float)cnt;
    const float m   = st / fmaxf(fc, 1.f);
    const float var = (sst - fc * m * m) / fmaxf(fc - 1.f, 1.f);
    meanw[k * C_DIM + c] = m;
    rstdw[k * C_DIM + c] = rsqrtf(EPSV + var);
  }
}

// ---------------- 5. per-class Gram via WMMA ----------------
// One workgroup per class; 8 waves; wave w owns output tile-row m=w (8 tiles of
// 16x16). G^T G is built from 64-row LDS-staged chunks of normalized f16 rows.
// NOTE: the A fragment is loaded into its own named variable (fragA) so that
// no register array is ever dynamically indexed -> no scratch spills.
__global__ __launch_bounds__(256)
void k_gemm(const float* __restrict__ x, const int* __restrict__ idxbuf,
            const int* __restrict__ counts, const float* __restrict__ meanw,
            const float* __restrict__ rstdw, float* __restrict__ classOff) {
  const int k = blockIdx.x;
  const int t = threadIdx.x;
  const int cnt = counts[k];
  if (cnt <= 1) {                         // invalid class -> contributes 0
    if (t == 0) classOff[k] = 0.f;
    return;
  }
  const int P = (cnt < PADG) ? cnt : PADG;

  const int lane = t & 31;
  const int wave = t >> 5;                // 0..7
  const int c    = t & (C_DIM - 1);       // staging column (fixed per thread)
  const int rh   = t >> 7;                // staging row interleave 0/1
  const int half16 = (lane >= 16) ? 8 : 0;
  const int mrow   = lane & 15;

  __shared__ __attribute__((aligned(16))) _Float16 stage[C_DIM * LDS_STRIDE];

  const float gm = meanw[k * C_DIM + c];
  const float gr = rstdw[k * C_DIM + c];

  // Base pointers inside the staged tile (runtime addresses are fine; only
  // register arrays must avoid runtime indices).
  const _Float16* rowA = &stage[(wave * 16 + mrow) * LDS_STRIDE + half16];

  v8f acc[8] = {};

  const int nchunks = (P + 63) >> 6;
  for (int ch = 0; ch < nchunks; ++ch) {
    const int pbase = ch << 6;
    // ---- stage 64 normalized rows, transposed [c][r] ----
    #pragma unroll
    for (int i = 0; i < 32; ++i) {
      const int r = 2 * i + rh;
      const int p = pbase + r;
      float v = 0.f;
      if (p < P) {
        if (p + 64 < P) __builtin_prefetch(&idxbuf[k * PADIDX + p + 64], 0, 1);
        const int row = idxbuf[k * PADIDX + p];
        v = (x[row * C_DIM + c] - gm) * gr;
      }
      stage[c * LDS_STRIDE + r] = (_Float16)v;
    }
    __syncthreads();
    // ---- 2 x K=32 WMMA steps over the 64-row chunk ----
    #pragma unroll
    for (int ks = 0; ks < 2; ++ks) {
      const int kb = ks * 32;
      // A fragment for this wave's tile row (own ds_load, no array indexing)
      const v8h alo = *(const v8h*)(rowA + kb);
      const v8h ahi = *(const v8h*)(rowA + kb + 16);
      const v16h fragA = __builtin_shufflevector(alo, ahi,
                    0, 1, 2, 3, 4, 5, 6, 7, 8, 9, 10, 11, 12, 13, 14, 15);
      // B fragments: constant-indexed only -> stay in VGPRs
      v16h fragB[8];
      #pragma unroll
      for (int n = 0; n < 8; ++n) {
        // 16-bit matrix layout: lanes 0-15 take K=kb+{0..7,16..23},
        // lanes 16-31 take +8; each half is one contiguous ds_load_b128.
        const _Float16* pr = &stage[(n * 16 + mrow) * LDS_STRIDE + kb + half16];
        const v8h lo = *(const v8h*)pr;
        const v8h hi = *(const v8h*)(pr + 16);
        fragB[n] = __builtin_shufflevector(lo, hi,
                    0, 1, 2, 3, 4, 5, 6, 7, 8, 9, 10, 11, 12, 13, 14, 15);
      }
      #pragma unroll
      for (int n = 0; n < 8; ++n) {
        acc[n] = __builtin_amdgcn_wmma_f32_16x16x32_f16(
            /*neg_a=*/false, fragA, /*neg_b=*/false, fragB[n],
            /*c_mod=*/(short)0, acc[n], /*reuse_a=*/false, /*reuse_b=*/false);
      }
    }
    __syncthreads();
  }

  // ---- off-diagonal^2 partial sum; C/D layout: VGPR r -> M=w*16+r+half16 ----
  float partial = 0.f;
  #pragma unroll
  for (int n = 0; n < 8; ++n) {
    const int N = n * 16 + mrow;
    #pragma unroll
    for (int r = 0; r < 8; ++r) {
      const int M = wave * 16 + r + half16;
      const float v = acc[n][r];
      if (M != N) partial += v * v;
    }
  }
  __shared__ float red[256];
  red[t] = partial;
  __syncthreads();
  for (int s = 128; s > 0; s >>= 1) {
    if (t < s) red[t] += red[t + s];
    __syncthreads();
  }
  if (t == 0) classOff[k] = red[0] / ((float)C_DIM * (float)(C_DIM - 1));
}

// ---------------- 6. final fixed-order reduction ----------------
__global__ __launch_bounds__(32)
void k_final(const int* __restrict__ counts, const float* __restrict__ classOff,
             float* __restrict__ out) {
  if (threadIdx.x != 0) return;
  float num = 0.f, nc = 0.f;
  for (int k = 0; k < KCLS; ++k) {
    if (counts[k] > 1) { num += classOff[k]; nc += (float)counts[k]; }
  }
  out[0] = (nc > 0.f) ? (num / nc) : 0.f;
}

// ---------------------------------------------------------------------------
extern "C" void kernel_launch(void* const* d_in, const int* in_sizes, int n_in,
                              void* d_out, int out_size, void* d_ws, size_t ws_size,
                              hipStream_t stream) {
  (void)in_sizes; (void)n_in; (void)out_size; (void)ws_size;
  const float* x = (const float*)d_in[0];
  const int*   y = (const int*)d_in[1];
  float* out = (float*)d_out;

  // workspace layout (every region read is written earlier in the same call)
  char* w = (char*)d_ws;
  size_t off = 0;
  auto take = [&](size_t bytes) { char* p = w + off; off = (off + bytes + 255) & ~(size_t)255; return p; };
  int*   counts      = (int*)  take(KCLS * sizeof(int));
  int*   blockHist   = (int*)  take((size_t)NB_HIST * KCLS * sizeof(int));
  int*   blockOffset = (int*)  take((size_t)NB_HIST * KCLS * sizeof(int));
  int*   idxbuf      = (int*)  take((size_t)KCLS * PADIDX * sizeof(int));
  float* meanw       = (float*)take((size_t)KCLS * C_DIM * sizeof(float));
  float* rstdw       = (float*)take((size_t)KCLS * C_DIM * sizeof(float));
  float* classOff    = (float*)take(KCLS * sizeof(float));

  k_hist   <<<NB_HIST, 256, 0, stream>>>(y, blockHist);
  k_scan   <<<1,       128, 0, stream>>>(blockHist, blockOffset, counts);
  k_scatter<<<NB_HIST, 256, 0, stream>>>(y, blockOffset, idxbuf);
  k_stats  <<<KCLS,    256, 0, stream>>>(x, idxbuf, counts, meanw, rstdw);
  k_gemm   <<<KCLS,    256, 0, stream>>>(x, idxbuf, counts, meanw, rstdw, classOff);
  k_final  <<<1,        32, 0, stream>>>(counts, classOff, out);
}
